// HMTE_73478300499971
// MI455X (gfx1250) — compile-verified, matrix operand order checked by reading, not run on hardware
//
#include <hip/hip_runtime.h>
#include <hip/hip_bf16.h>
#include <math.h>

// ---------------------------------------------------------------------------
// HMTE forward for MI455X (gfx1250, wave32, WMMA).
// Heavy lifting = bf16 WMMA GEMMs (v_wmma_f32_16x16x32_bf16), everything else
// is small VALU work. Final scoring GEMM (1024 x 100000 x 256) dominates and
// is output-bandwidth bound (~0.46 GB traffic -> ~20us at 23.3 TB/s); its
// 409 MB f32 output is emitted with non-temporal stores so the 51 MB bf16
// entity table stays resident in the 192 MB L2.
// ---------------------------------------------------------------------------

typedef __bf16 bhalf;
typedef __attribute__((ext_vector_type(16))) __bf16 bf16x16;
typedef __attribute__((ext_vector_type(8)))  __bf16 bf16x8;
typedef __attribute__((ext_vector_type(8)))  float  fx8;

#define BATCH   1024
#define SEQL    5
#define DMODEL  256
#define DINNER  512
#define DSTATE  16
#define NHEADS  8      // mamba heads
#define HEADD   64
#define CONVD   544    // DINNER + 2*DSTATE
#define ZXW     1064   // 2*DINNER + 2*DSTATE + NHEADS
#define ROWS    (BATCH * SEQL)      // 5120
#define NENT    100000
#define AHEADS  16     // attention heads
#define AE      16     // attention head dim

__device__ __forceinline__ float siluf(float x) {
    return x / (1.0f + expf(-x));
}

__device__ __forceinline__ float wave_sum(float v) {
#pragma unroll
    for (int o = 16; o > 0; o >>= 1) v += __shfl_xor(v, o, 32);
    return v;
}

// ---------------------------------------------------------------------------
// Generic GEMM: C[M,N] = A[M,K](bf16,row-major) * Bt[N,K](bf16,row-major)^T
//               (+ bias[N] if non-null), f32 accumulate via WMMA.
// Requirements: M % 32 == 0, K % 32 == 0 (K is a template constant: full
// unroll -> immediate-offset b128 loads, scheduler pipelines loads past
// WMMAs). N is guarded: out-of-range lanes load a clamped B row; in the
// 16-bit B fragment layout column n's data lives only in lanes {n, n+16},
// so the garbage only reaches output columns that are never stored. This
// keeps EXEC uniform (required by WMMA) with no per-iteration zero-fill.
//
// bf16 A-fragment layout (CDNA5 ISA 7.12.2, 16-bit A 16x32):
//   lane l: m = l&15, half = l>>4
//   element e in [0,8):  k = 8*half + e          (16 contiguous bytes)
//   element e in [8,16): k = 16 + 8*half + (e-8) (16 contiguous bytes)
// B uses the mirrored layout with n = l&15 over Bt rows (Bt[n][k] = B[k][n]).
// C/D: vgpr r holds row m = 8*(l>>4) + r, col n = l&15.
// ---------------------------------------------------------------------------
template <int K, bool NTSTORE>
__global__ __launch_bounds__(256) void k_gemm_bf16_wmma(
    const bhalf* __restrict__ A, const bhalf* __restrict__ Bt,
    const float* __restrict__ bias, float* __restrict__ C,
    int M, int N)
{
    const int lane = threadIdx.x & 31;
    const int wave = threadIdx.x >> 5;
    const int half = lane >> 4;
    const int l15  = lane & 15;

    const int m0 = blockIdx.y * 32 + (wave & 1) * 16;
    const int n0 = blockIdx.x * 64 + (wave >> 1) * 16;
    const int m  = m0 + l15;
    const int n  = n0 + l15;
    const bool nvalid = (n < N);
    const int  nc = nvalid ? n : (N - 1);   // clamped row; garbage cols dropped

    fx8 acc;
#pragma unroll
    for (int i = 0; i < 8; ++i) acc[i] = 0.0f;

    const bhalf* ap = A + (size_t)m * K + 8 * half;
    const bhalf* bp = Bt + (size_t)nc * K + 8 * half;

    // one-shot L2 prefetch of the fragment rows (global_prefetch_b8)
    __builtin_prefetch(ap, 0, 3);
    __builtin_prefetch(bp, 0, 3);

#pragma unroll
    for (int k0 = 0; k0 < K; k0 += 32) {
        bf16x8 alo = *reinterpret_cast<const bf16x8*>(ap + k0);
        bf16x8 ahi = *reinterpret_cast<const bf16x8*>(ap + k0 + 16);
        bf16x8 blo = *reinterpret_cast<const bf16x8*>(bp + k0);
        bf16x8 bhi = *reinterpret_cast<const bf16x8*>(bp + k0 + 16);
        bf16x16 afrag, bfrag;
#pragma unroll
        for (int i = 0; i < 8; ++i) {
            afrag[i] = alo[i]; afrag[i + 8] = ahi[i];
            bfrag[i] = blo[i]; bfrag[i + 8] = bhi[i];
        }
        acc = __builtin_amdgcn_wmma_f32_16x16x32_bf16(
            false, afrag, false, bfrag, (short)0, acc, false, false);
    }

    if (!nvalid) return;
    const float bv = bias ? bias[n] : 0.0f;
#pragma unroll
    for (int r = 0; r < 8; ++r) {
        int mm = m0 + 8 * half + r;
        float v = acc[r] + bv;
        if (NTSTORE)
            __builtin_nontemporal_store(v, &C[(size_t)mm * N + n]);
        else
            C[(size_t)mm * N + n] = v;
    }
}

// ---------------------------------------------------------------------------
// Weight prep: W[K,N] f32 (row-major) -> Wt[N,K] bf16 (row-major)
// ---------------------------------------------------------------------------
__global__ void k_transpose_bf16(const float* __restrict__ W, bhalf* __restrict__ Wt,
                                 int K, int N)
{
    int id = blockIdx.x * 256 + threadIdx.x;
    if (id >= K * N) return;
    int k = id % K;
    int n = id / K;
    Wt[id] = (bhalf)W[(size_t)k * N + n];
}

// ---------------------------------------------------------------------------
// Build x0[b,t,:] = (t==0 ? rel_e[rel_idx[b]] : ent_e[ent_idx[b][t-1]]) + pos_e[p]
// with p = t < miss ? t : t+1. Writes f32 and bf16 copies.
// ---------------------------------------------------------------------------
__global__ void k_gather_embed(const int* __restrict__ rel_idx, const int* __restrict__ ent_idx,
                               const int* __restrict__ miss_p,
                               const float* __restrict__ ent_e, const float* __restrict__ rel_e,
                               const float* __restrict__ pos_e,
                               float* __restrict__ xf, bhalf* __restrict__ xb)
{
    int id = blockIdx.x * 256 + threadIdx.x;
    if (id >= ROWS * DMODEL) return;
    int d = id & (DMODEL - 1);
    int t = (id >> 8) % SEQL;
    int b = id / (SEQL * DMODEL);
    int miss = *miss_p;
    int p = (t < miss) ? t : t + 1;
    float v;
    if (t == 0) v = rel_e[(size_t)rel_idx[b] * DMODEL + d];
    else        v = ent_e[(size_t)ent_idx[b * 4 + (t - 1)] * DMODEL + d];
    v += pos_e[p * DMODEL + d];
    xf[id] = v;
    xb[id] = (bhalf)v;
}

// ---------------------------------------------------------------------------
// Depthwise causal conv (k=4, pad 3) over L for xBC channels + SiLU.
// zx: [ROWS, ZXW]; xBC slice = cols [DINNER, DINNER+CONVD)
// ---------------------------------------------------------------------------
__global__ void k_conv_silu(const float* __restrict__ zx, const float* __restrict__ cw,
                            const float* __restrict__ cb, float* __restrict__ out)
{
    int id = blockIdx.x * 256 + threadIdx.x;
    if (id >= BATCH * CONVD) return;
    int c = id % CONVD;
    int b = id / CONVD;
    float w[4];
#pragma unroll
    for (int j = 0; j < 4; ++j) w[j] = cw[c * 4 + j];
    float bias = cb[c];
    float xin[SEQL];
#pragma unroll
    for (int t = 0; t < SEQL; ++t)
        xin[t] = zx[(size_t)(b * SEQL + t) * ZXW + DINNER + c];
#pragma unroll
    for (int t = 0; t < SEQL; ++t) {
        float acc = bias;
#pragma unroll
        for (int j = 0; j < 4; ++j) {
            int tt = t - 3 + j;
            if (tt >= 0) acc += w[j] * xin[tt];
        }
        out[(size_t)(b * SEQL + t) * CONVD + c] = siluf(acc);
    }
}

// ---------------------------------------------------------------------------
// dt = softplus(raw + dt_bias); dA = exp(-dt * exp(A_log)), per (row, head)
// ---------------------------------------------------------------------------
__global__ void k_dt(const float* __restrict__ zx, const float* __restrict__ dt_bias,
                     const float* __restrict__ A_log,
                     float* __restrict__ dtv, float* __restrict__ dAv)
{
    int id = blockIdx.x * 256 + threadIdx.x;
    if (id >= ROWS * NHEADS) return;
    int h = id & (NHEADS - 1);
    int row = id >> 3;
    float r = zx[(size_t)row * ZXW + (ZXW - NHEADS) + h] + dt_bias[h];
    float sp = (r > 20.0f) ? r : log1pf(expf(r));
    dtv[id] = sp;
    dAv[id] = expf(-sp * expf(A_log[h]));
}

// ---------------------------------------------------------------------------
// SSM scan: thread per (b, h, p) holds 16 states in registers, walks L=5.
// y[row, h*64+p] = sum_n h_state[n]*C[n] + D[h]*x
// ---------------------------------------------------------------------------
__global__ void k_scan(const float* __restrict__ xBC, const float* __restrict__ dtv,
                       const float* __restrict__ dAv, const float* __restrict__ Dp,
                       float* __restrict__ y)
{
    int id = blockIdx.x * 256 + threadIdx.x;
    if (id >= BATCH * DINNER) return;
    int p = id & (HEADD - 1);
    int h = (id >> 6) & (NHEADS - 1);
    int b = id >> 9;
    float st[DSTATE];
#pragma unroll
    for (int n = 0; n < DSTATE; ++n) st[n] = 0.0f;
    float Dh = Dp[h];
#pragma unroll
    for (int t = 0; t < SEQL; ++t) {
        size_t row = (size_t)b * SEQL + t;
        float dA = dAv[row * NHEADS + h];
        float dt = dtv[row * NHEADS + h];
        float xv = xBC[row * CONVD + h * HEADD + p];
        const float* Bp = &xBC[row * CONVD + DINNER];
        const float* Cp = Bp + DSTATE;
        float dx = dt * xv;
        float acc = 0.0f;
#pragma unroll
        for (int n = 0; n < DSTATE; ++n) {
            st[n] = st[n] * dA + dx * Bp[n];
            acc += st[n] * Cp[n];
        }
        y[row * DINNER + h * HEADD + p] = acc + Dh * xv;
    }
}

// ---------------------------------------------------------------------------
// y <- y * silu(z); RMS-norm over 512 with norm_w; emit bf16 for out_proj GEMM.
// One wave per row (lane owns 16 strided elements), wave32 shfl reduction.
// ---------------------------------------------------------------------------
__global__ __launch_bounds__(256) void k_gate_rms(const float* __restrict__ y,
                                                  const float* __restrict__ zx,
                                                  const float* __restrict__ norm_w,
                                                  bhalf* __restrict__ out)
{
    int lane = threadIdx.x & 31;
    int row = blockIdx.x * 8 + (threadIdx.x >> 5);
    float g[16];
    float ss = 0.0f;
#pragma unroll
    for (int j = 0; j < 16; ++j) {
        int i = lane + 32 * j;
        float z = zx[(size_t)row * ZXW + i];
        float v = y[(size_t)row * DINNER + i] * siluf(z);
        g[j] = v;
        ss += v * v;
    }
    ss = wave_sum(ss);
    float inv = rsqrtf(ss * (1.0f / DINNER) + 1e-5f);
#pragma unroll
    for (int j = 0; j < 16; ++j) {
        int i = lane + 32 * j;
        out[(size_t)row * DINNER + i] = (bhalf)(g[j] * inv * norm_w[i]);
    }
}

// ---------------------------------------------------------------------------
// x <- LayerNorm(x + add) * w + b ; also write bf16 copy for the next GEMM.
// One wave per 256-wide row.
// ---------------------------------------------------------------------------
__global__ __launch_bounds__(256) void k_residual_ln(float* __restrict__ x,
                                                     const float* __restrict__ add,
                                                     const float* __restrict__ w,
                                                     const float* __restrict__ bb,
                                                     bhalf* __restrict__ xb)
{
    int lane = threadIdx.x & 31;
    int row = blockIdx.x * 8 + (threadIdx.x >> 5);
    float v[8];
    float s = 0.0f, s2 = 0.0f;
#pragma unroll
    for (int j = 0; j < 8; ++j) {
        int i = lane + 32 * j;
        float t = x[(size_t)row * DMODEL + i] + add[(size_t)row * DMODEL + i];
        v[j] = t;
        s += t;
        s2 += t * t;
    }
    s = wave_sum(s);
    s2 = wave_sum(s2);
    float mean = s * (1.0f / DMODEL);
    float var = s2 * (1.0f / DMODEL) - mean * mean;
    float inv = rsqrtf(var + 1e-5f);
#pragma unroll
    for (int j = 0; j < 8; ++j) {
        int i = lane + 32 * j;
        float o = (v[j] - mean) * inv * w[i] + bb[i];
        x[(size_t)row * DMODEL + i] = o;
        xb[(size_t)row * DMODEL + i] = (bhalf)o;
    }
}

// ---------------------------------------------------------------------------
// Tiny causal attention (L=5, 16 heads x 16 dim). Thread per (b, head, l).
// Writes bf16 so Wo projection can consume directly.
// ---------------------------------------------------------------------------
__global__ void k_attn(const float* __restrict__ q, const float* __restrict__ k,
                       const float* __restrict__ v, bhalf* __restrict__ o)
{
    int id = blockIdx.x * 256 + threadIdx.x;
    if (id >= BATCH * AHEADS * SEQL) return;
    int l = id % SEQL;
    int h = (id / SEQL) & (AHEADS - 1);
    int b = id / (SEQL * AHEADS);
    size_t base = (size_t)(b * SEQL) * DMODEL + h * AE;
    float qv[AE];
#pragma unroll
    for (int e = 0; e < AE; ++e) qv[e] = q[base + (size_t)l * DMODEL + e];
    float sc[SEQL];
    float mx = -1e30f;
    for (int s = 0; s <= l; ++s) {
        float d = 0.0f;
#pragma unroll
        for (int e = 0; e < AE; ++e) d += qv[e] * k[base + (size_t)s * DMODEL + e];
        d *= 0.25f; // 1/sqrt(16)
        sc[s] = d;
        mx = fmaxf(mx, d);
    }
    float denom = 0.0f;
    for (int s = 0; s <= l; ++s) { sc[s] = expf(sc[s] - mx); denom += sc[s]; }
    float invd = 1.0f / denom;
    float ov[AE];
#pragma unroll
    for (int e = 0; e < AE; ++e) ov[e] = 0.0f;
    for (int s = 0; s <= l; ++s) {
#pragma unroll
        for (int e = 0; e < AE; ++e) ov[e] += sc[s] * v[base + (size_t)s * DMODEL + e];
    }
#pragma unroll
    for (int e = 0; e < AE; ++e)
        o[base + (size_t)l * DMODEL + e] = (bhalf)(ov[e] * invd);
}

// x[:, L-1, :] -> bf16 (1024 x 256)
__global__ void k_last_bf16(const float* __restrict__ x, bhalf* __restrict__ xl)
{
    int id = blockIdx.x * 256 + threadIdx.x;
    if (id >= BATCH * DMODEL) return;
    int d = id & (DMODEL - 1);
    int b = id >> 8;
    xl[id] = (bhalf)x[(size_t)(b * SEQL + (SEQL - 1)) * DMODEL + d];
}

// tar[e,d] = ent_e[e,d] + pos_e[miss,d] -> bf16 (100000 x 256)
__global__ void k_tar(const float* __restrict__ ent_e, const float* __restrict__ pos_e,
                      const int* __restrict__ miss_p, bhalf* __restrict__ tar)
{
    size_t id = (size_t)blockIdx.x * 256 + threadIdx.x;
    if (id >= (size_t)NENT * DMODEL) return;
    int d = (int)(id & (DMODEL - 1));
    tar[id] = (bhalf)(ent_e[id] + pos_e[(*miss_p) * DMODEL + d]);
}

// ---------------------------------------------------------------------------
// Host-side orchestration
// ---------------------------------------------------------------------------
namespace {

struct MambaW {
    const float *in_proj, *conv_w, *conv_b, *dt_bias, *A_log, *Dp, *norm_w, *out_proj;
};

struct Ws {
    float* x_f32;     // ROWS x DMODEL
    bhalf* x_bf16;    // ROWS x DMODEL
    bhalf* wT;        // up to 1064*256 bf16 (reused for every weight)
    float* zx;        // ROWS x ZXW   (reused as q/k/v during attention)
    float* xBCc;      // ROWS x CONVD
    float* dtv;       // ROWS x NHEADS
    float* dAv;       // ROWS x NHEADS
    float* ybuf;      // ROWS x DINNER
    bhalf* ybar;      // ROWS x DINNER bf16
    float* gout;      // ROWS x DMODEL (mamba/attn block output)
    bhalf* o_bf16;    // ROWS x DMODEL bf16
    bhalf* xl;        // BATCH x DMODEL bf16
    bhalf* tar;       // NENT x DMODEL bf16
};

inline int ceil_div(int a, int b) { return (a + b - 1) / b; }

void launch_gemm(const bhalf* A, const bhalf* Bt, const float* bias, float* C,
                 int M, int N, int K, bool nt, hipStream_t s)
{
    dim3 g(ceil_div(N, 64), M / 32);
    if (K == 256) {
        if (nt) k_gemm_bf16_wmma<256, true ><<<g, 256, 0, s>>>(A, Bt, bias, C, M, N);
        else    k_gemm_bf16_wmma<256, false><<<g, 256, 0, s>>>(A, Bt, bias, C, M, N);
    } else { // K == 512
        if (nt) k_gemm_bf16_wmma<512, true ><<<g, 256, 0, s>>>(A, Bt, bias, C, M, N);
        else    k_gemm_bf16_wmma<512, false><<<g, 256, 0, s>>>(A, Bt, bias, C, M, N);
    }
}

void launch_transpose(const float* W, bhalf* Wt, int K, int N, hipStream_t s)
{
    k_transpose_bf16<<<ceil_div(K * N, 256), 256, 0, s>>>(W, Wt, K, N);
}

void run_mamba(const MambaW& w, const Ws& b, hipStream_t s)
{
    // zxbcdt = x @ in_proj     (5120 x 1064 x 256)
    launch_transpose(w.in_proj, b.wT, DMODEL, ZXW, s);
    launch_gemm(b.x_bf16, b.wT, nullptr, b.zx, ROWS, ZXW, DMODEL, false, s);
    // causal depthwise conv + silu
    k_conv_silu<<<ceil_div(BATCH * CONVD, 256), 256, 0, s>>>(b.zx, w.conv_w, w.conv_b, b.xBCc);
    // dt / dA
    k_dt<<<ceil_div(ROWS * NHEADS, 256), 256, 0, s>>>(b.zx, w.dt_bias, w.A_log, b.dtv, b.dAv);
    // selective scan
    k_scan<<<ceil_div(BATCH * DINNER, 256), 256, 0, s>>>(b.xBCc, b.dtv, b.dAv, w.Dp, b.ybuf);
    // gate by silu(z), RMS-norm, emit bf16
    k_gate_rms<<<ROWS / 8, 256, 0, s>>>(b.ybuf, b.zx, w.norm_w, b.ybar);
    // out = ybar @ out_proj    (5120 x 256 x 512)
    launch_transpose(w.out_proj, b.wT, DINNER, DMODEL, s);
    launch_gemm(b.ybar, b.wT, nullptr, b.gout, ROWS, DMODEL, DINNER, false, s);
}

} // namespace

extern "C" void kernel_launch(void* const* d_in, const int* in_sizes, int n_in,
                              void* d_out, int out_size, void* d_ws, size_t ws_size,
                              hipStream_t stream)
{
    // ---- inputs (setup_inputs() insertion order, nested dicts flattened) ----
    const int*   rel_idx = (const int*)d_in[0];
    const int*   ent_idx = (const int*)d_in[1];
    const int*   miss_p  = (const int*)d_in[2];
    const float* ent_e   = (const float*)d_in[3];
    const float* rel_e   = (const float*)d_in[4];
    const float* pos_e   = (const float*)d_in[5];
    const float* bvec    = (const float*)d_in[6];

    MambaW pre_mamba = {
        (const float*)d_in[7],  (const float*)d_in[8],  (const float*)d_in[9],
        (const float*)d_in[10], (const float*)d_in[11], (const float*)d_in[12],
        (const float*)d_in[13], (const float*)d_in[14]
    };
    const float* preln_w = (const float*)d_in[15];
    const float* preln_b = (const float*)d_in[16];

    // ---- workspace carve-out (deterministic bump allocator) ----
    char* p = (char*)d_ws;
    auto alloc = [&](size_t bytes) -> void* {
        void* r = (void*)p;
        p += (bytes + 255) & ~(size_t)255;
        return r;
    };
    Ws b;
    b.x_f32  = (float*)alloc((size_t)ROWS * DMODEL * 4);
    b.x_bf16 = (bhalf*)alloc((size_t)ROWS * DMODEL * 2);
    b.wT     = (bhalf*)alloc((size_t)ZXW * DMODEL * 2);
    b.zx     = (float*)alloc((size_t)ROWS * ZXW * 4);
    b.xBCc   = (float*)alloc((size_t)ROWS * CONVD * 4);
    b.dtv    = (float*)alloc((size_t)ROWS * NHEADS * 4);
    b.dAv    = (float*)alloc((size_t)ROWS * NHEADS * 4);
    b.ybuf   = (float*)alloc((size_t)ROWS * DINNER * 4);
    b.ybar   = (bhalf*)alloc((size_t)ROWS * DINNER * 2);
    b.gout   = (float*)alloc((size_t)ROWS * DMODEL * 4);
    b.o_bf16 = (bhalf*)alloc((size_t)ROWS * DMODEL * 2);
    b.xl     = (bhalf*)alloc((size_t)BATCH * DMODEL * 2);
    b.tar    = (bhalf*)alloc((size_t)NENT * DMODEL * 2);

    // q/k/v reuse the zx region (needs 3*ROWS*DMODEL < ROWS*ZXW floats)
    float* qbuf = b.zx;
    float* kbuf = b.zx + (size_t)ROWS * DMODEL;
    float* vbuf = b.zx + (size_t)2 * ROWS * DMODEL;

    // ---- 1) build x0 = [rel ; ents] + pos ----
    k_gather_embed<<<ceil_div(ROWS * DMODEL, 256), 256, 0, stream>>>(
        rel_idx, ent_idx, miss_p, ent_e, rel_e, pos_e, b.x_f32, b.x_bf16);

    // ---- 2) pre-mamba + residual LN ----
    run_mamba(pre_mamba, b, stream);
    k_residual_ln<<<ROWS / 8, 256, 0, stream>>>(b.x_f32, b.gout, preln_w, preln_b, b.x_bf16);

    // ---- 3) attention+mamba layers ----
    for (int layer = 0; layer < 2; ++layer) {
        const int L = 17 + layer * 20;
        const float* Wq = (const float*)d_in[L + 0];
        const float* bq = (const float*)d_in[L + 1];
        const float* Wk = (const float*)d_in[L + 2];
        const float* bk = (const float*)d_in[L + 3];
        const float* Wv = (const float*)d_in[L + 4];
        const float* bv = (const float*)d_in[L + 5];
        const float* Wo = (const float*)d_in[L + 6];
        const float* bo = (const float*)d_in[L + 7];
        const float* ln1_w = (const float*)d_in[L + 8];
        const float* ln1_b = (const float*)d_in[L + 9];
        const float* ln2_w = (const float*)d_in[L + 10];
        const float* ln2_b = (const float*)d_in[L + 11];
        MambaW mw = {
            (const float*)d_in[L + 12], (const float*)d_in[L + 13],
            (const float*)d_in[L + 14], (const float*)d_in[L + 15],
            (const float*)d_in[L + 16], (const float*)d_in[L + 17],
            (const float*)d_in[L + 18], (const float*)d_in[L + 19]
        };

        // q/k/v projections
        launch_transpose(Wq, b.wT, DMODEL, DMODEL, stream);
        launch_gemm(b.x_bf16, b.wT, bq, qbuf, ROWS, DMODEL, DMODEL, false, stream);
        launch_transpose(Wk, b.wT, DMODEL, DMODEL, stream);
        launch_gemm(b.x_bf16, b.wT, bk, kbuf, ROWS, DMODEL, DMODEL, false, stream);
        launch_transpose(Wv, b.wT, DMODEL, DMODEL, stream);
        launch_gemm(b.x_bf16, b.wT, bv, vbuf, ROWS, DMODEL, DMODEL, false, stream);

        // causal softmax attention
        k_attn<<<ceil_div(BATCH * AHEADS * SEQL, 256), 256, 0, stream>>>(
            qbuf, kbuf, vbuf, b.o_bf16);

        // output projection + residual LN1
        launch_transpose(Wo, b.wT, DMODEL, DMODEL, stream);
        launch_gemm(b.o_bf16, b.wT, bo, b.gout, ROWS, DMODEL, DMODEL, false, stream);
        k_residual_ln<<<ROWS / 8, 256, 0, stream>>>(b.x_f32, b.gout, ln1_w, ln1_b, b.x_bf16);

        // mamba block + residual LN2
        run_mamba(mw, b, stream);
        k_residual_ln<<<ROWS / 8, 256, 0, stream>>>(b.x_f32, b.gout, ln2_w, ln2_b, b.x_bf16);
    }

    // ---- 4) scoring GEMM: out = x_last @ (ent_e + pos_e[miss])^T + b ----
    // Non-temporal output stores: 409 MB write-once stream must not rinse L2.
    k_last_bf16<<<ceil_div(BATCH * DMODEL, 256), 256, 0, stream>>>(b.x_f32, b.xl);
    k_tar<<<ceil_div(NENT * DMODEL, 256), 256, 0, stream>>>(ent_e, pos_e, miss_p, b.tar);
    launch_gemm(b.xl, b.tar, bvec, (float*)d_out, BATCH, NENT, DMODEL, true, stream);

    (void)in_sizes; (void)n_in; (void)out_size; (void)ws_size;
}
